// NoisyTopkRouter_43482248904859
// MI455X (gfx1250) — matrix operand order repaired and checked by reading, not compiled, not run on hardware
//
#include <hip/hip_runtime.h>

// ---------------------------------------------------------------------------
// NoisyTopkRouter for MI455X (gfx1250, wave32, WMMA).
//   Pass 1: convert W_route / W_noise fp32 -> bf16 into d_ws (512 KB, L2-hot).
//   Pass 2: x[32768,2048] @ W^T via v_wmma_f32_16x16x32_bf16, f32 accum.
//           B staged per 32-K chunk in LDS (double-buffered, fragment layout)
//           so weights cross L2 once per *block*, not once per *wave* (8x cut:
//           ~1 GB -> ~128 MB of L2 traffic).
//           Epilogue: noisy = route + eps*softplus(noise); top-2; 2-way
//           softmax scatter (all other experts exactly 0); indices.
// Roofline: x read-once (268 MB) -> ~12 us @ 23.3 TB/s. bf16 WMMA keeps the
// 17.2 GFLOP (needs ~1.4 PFLOP/s effective) off the critical path.
// ---------------------------------------------------------------------------

typedef __attribute__((ext_vector_type(16))) __bf16 v16bf;
typedef __attribute__((ext_vector_type(4)))  __bf16 v4bf;
typedef __attribute__((ext_vector_type(8)))  float  v8f;
typedef __attribute__((ext_vector_type(4)))  float  v4f;

static constexpr int kD      = 2048;      // reduction dim
static constexpr int kE      = 64;        // experts
static constexpr int kRows   = 8 * 4096;
static constexpr int kMTile  = 128;       // rows per block
static constexpr int kKC     = 32;        // K per WMMA
static constexpr int kKSteps = kD / kKC;  // 64
static constexpr int kWElems = kE * kD;   // elements per weight matrix

// ---------------------------------------------------------------------------
// Pass 1: fp32 -> bf16 weight conversion (both matrices), 4 elems/thread.
// ---------------------------------------------------------------------------
__global__ __launch_bounds__(256)
void convert_weights_kernel(const float* __restrict__ Wr,
                            const float* __restrict__ Wn,
                            __bf16* __restrict__ out) {
  const int q = (blockIdx.x * blockDim.x + threadIdx.x) * 4;
  if (q >= 2 * kWElems) return;
  const float* src = (q < kWElems) ? (Wr + q) : (Wn + (q - kWElems));
  v4f v = *(const v4f*)src;
  v4bf o;
  o[0] = (__bf16)v[0]; o[1] = (__bf16)v[1];
  o[2] = (__bf16)v[2]; o[3] = (__bf16)v[3];
  *(v4bf*)(out + q) = o;
}

// Pack 8 floats at p0 -> elements 0..7, 8 floats at p1 -> elements 8..15.
__device__ __forceinline__ v16bf pack_bf16(const float* p0, const float* p1) {
  v4f a = *(const v4f*)(p0);
  v4f b = *(const v4f*)(p0 + 4);
  v4f c = *(const v4f*)(p1);
  v4f d = *(const v4f*)(p1 + 4);
  v16bf r;
  r[0]  = (__bf16)a[0]; r[1]  = (__bf16)a[1]; r[2]  = (__bf16)a[2]; r[3]  = (__bf16)a[3];
  r[4]  = (__bf16)b[0]; r[5]  = (__bf16)b[1]; r[6]  = (__bf16)b[2]; r[7]  = (__bf16)b[3];
  r[8]  = (__bf16)c[0]; r[9]  = (__bf16)c[1]; r[10] = (__bf16)c[2]; r[11] = (__bf16)c[3];
  r[12] = (__bf16)d[0]; r[13] = (__bf16)d[1]; r[14] = (__bf16)d[2]; r[15] = (__bf16)d[3];
  return r;
}

__device__ __forceinline__ float softplusf(float z) {
  // softplus(z) = max(z,0) + log1p(exp(-|z|))  (stable)
  return fmaxf(z, 0.0f) + log1pf(__expf(-fabsf(z)));
}

// ---------------------------------------------------------------------------
// Pass 2: GEMM + noisy top-2 router.
// ---------------------------------------------------------------------------
__global__ __launch_bounds__(256, 2)
void noisy_topk_router_kernel(const float*  __restrict__ x,
                              const __bf16* __restrict__ Wrb,   // bf16 [E][D]
                              const __bf16* __restrict__ Wnb,   // bf16 [E][D]
                              const float*  __restrict__ br,
                              const float*  __restrict__ bn,
                              const float*  __restrict__ eps,
                              float* __restrict__ out_router,
                              int*   __restrict__ out_idx) {
  // Epilogue scratch (+1 float pad kills the 32-way bank conflict of
  // row-per-thread reads).
  __shared__ float smem[kMTile][2 * kE + 1];
  // B staging, WMMA fragment layout: [buf][frag][lane][16 bf16].
  // frag 0..3 = route col-tiles, frag 4..7 = noise col-tiles. 16 KB total.
  __shared__ alignas(32) __bf16 wlds[2][8][32][16];

  const int tid  = threadIdx.x;
  const int wave = tid >> 5;
  const int lane = tid & 31;
  const int half = lane >> 4;   // 0: lanes 0-15, 1: lanes 16-31
  const int l16  = lane & 15;

  const int rowBlk = blockIdx.x * kMTile;          // first row of this block
  const int rowA   = rowBlk + wave * 16 + l16;     // A-fragment row for lane

  const float* xrow = x + (size_t)rowA * kD;

  // Loader role: wave w fills fragment w. m = w>>2 (0=route,1=noise),
  // t = w&3 (col tile). Lane layout matches the consumer's B fragment:
  // lanes 0-15 hold K k0..k0+15 of expert t*16+l16, lanes 16-31 hold
  // K k0+16..k0+31 (B[k][n] = W[n][k] -> 32 contiguous bytes of one W row).
  const __bf16* wsrc =
      ((wave >> 2) ? Wnb : Wrb) + (size_t)((wave & 3) * 16 + l16) * kD + half * 16;

  v8f accR[4], accN[4];
  const v8f zero = {0.f, 0.f, 0.f, 0.f, 0.f, 0.f, 0.f, 0.f};
#pragma unroll
  for (int t = 0; t < 4; ++t) { accR[t] = zero; accN[t] = zero; }

  // Prologue: stage chunk 0, prefetch chunk 1 into registers.
  v16bf wnext;
  {
    v16bf w0 = *(const v16bf*)(wsrc);
    *(v16bf*)&wlds[0][wave][lane][0] = w0;
    wnext = *(const v16bf*)(wsrc + kKC);
  }
  __syncthreads();

  for (int kc = 0; kc < kKSteps; ++kc) {
    const int k0 = kc * kKC;
    // Speculative prefetch of next x chunk -> global_prefetch_b8
    __builtin_prefetch(xrow + k0 + kKC, 0, 0);

    // A fragment (16x32 bf16): lanes 0-15 hold K k0..k0+15,
    // lanes 16-31 hold K k0+16..k0+31.
    const int ka = k0 + half * 16;
    v16bf af = pack_bf16(xrow + ka, xrow + ka + 8);

    // B fragments come straight from LDS in fragment order: 2x ds_load_b128.
    const int buf = kc & 1;
#pragma unroll
    for (int t = 0; t < 4; ++t) {
      v16bf bfr = *(const v16bf*)&wlds[buf][t][lane][0];
      accR[t] = __builtin_amdgcn_wmma_f32_16x16x32_bf16(
          false, af, false, bfr, (short)0, accR[t], false, false);
      v16bf bfn = *(const v16bf*)&wlds[buf][4 + t][lane][0];
      accN[t] = __builtin_amdgcn_wmma_f32_16x16x32_bf16(
          false, af, false, bfn, (short)0, accN[t], false, false);
    }

    // Double-buffer handoff: stores into buf^1 are fenced from last
    // iteration's reads of buf^1 by the previous barrier; next iteration's
    // reads are fenced from these stores by the barrier below.
    if (kc + 1 < kKSteps) {
      *(v16bf*)&wlds[buf ^ 1][wave][lane][0] = wnext;
      const int knext = (kc + 2 < kKSteps) ? (kc + 2) * kKC : (kD - kKC);
      wnext = *(const v16bf*)(wsrc + knext);
    }
    __syncthreads();
  }

  // Spill accumulators: C/D layout is VGPR r -> M = r (lanes 0-15) or M = 8+r
  // (lanes 16-31); N = lane%16.
#pragma unroll
  for (int t = 0; t < 4; ++t) {
#pragma unroll
    for (int r = 0; r < 8; ++r) {
      const int rl = wave * 16 + r + half * 8;
      smem[rl][t * 16 + l16]       = accR[t][r];
      smem[rl][kE + t * 16 + l16]  = accN[t][r];
    }
  }
  __syncthreads();

  // Epilogue: one thread per row.
  if (tid < kMTile) {
    const size_t row = (size_t)rowBlk + tid;
    const float* ep  = eps + row * kE;

    float v1 = -INFINITY, v2 = -INFINITY;
    int   i1 = 0,         i2 = 0;
#pragma unroll 8
    for (int e = 0; e < kE; ++e) {
      const float lg    = smem[tid][e]      + br[e];
      const float nz    = smem[tid][kE + e] + bn[e];
      const float noisy = lg + ep[e] * softplusf(nz);
      if (noisy > v1)      { v2 = v1; i2 = i1; v1 = noisy; i1 = e; }
      else if (noisy > v2) { v2 = noisy; i2 = e; }
    }

    // Softmax over {v1, v2} with -inf background => everything else is 0.
    const float e2 = __expf(v2 - v1);       // v1 >= v2
    const float dn = 1.0f + e2;
    const float w1 = 1.0f / dn;
    const float w2 = e2 / dn;

    v4f* dst = (v4f*)(out_router + row * kE);
#pragma unroll
    for (int q = 0; q < kE / 4; ++q) {
      v4f v;
#pragma unroll
      for (int j = 0; j < 4; ++j) {
        const int e = q * 4 + j;
        v[j] = (e == i1) ? w1 : ((e == i2) ? w2 : 0.0f);
      }
      dst[q] = v;
    }
    out_idx[row * 2 + 0] = i1;
    out_idx[row * 2 + 1] = i2;
  }
}

extern "C" void kernel_launch(void* const* d_in, const int* in_sizes, int n_in,
                              void* d_out, int out_size, void* d_ws, size_t ws_size,
                              hipStream_t stream) {
  const float* x   = (const float*)d_in[0];  // [8,4096,2048]
  const float* Wr  = (const float*)d_in[1];  // [64,2048]
  const float* br  = (const float*)d_in[2];  // [64]
  const float* Wn  = (const float*)d_in[3];  // [64,2048]
  const float* bn  = (const float*)d_in[4];  // [64]
  const float* eps = (const float*)d_in[5];  // [8,4096,64]

  __bf16* Wrb = (__bf16*)d_ws;               // [64,2048] bf16
  __bf16* Wnb = Wrb + (size_t)kWElems;       // [64,2048] bf16

  float* out_router = (float*)d_out;                           // [8,4096,64]
  int*   out_idx    = (int*)(out_router + (size_t)kRows * kE); // [8,4096,2] i32

  // Pass 1: weight conversion (2*131072 elems, 4 per thread).
  const int cvt_threads = (2 * kWElems) / 4;
  convert_weights_kernel<<<dim3(cvt_threads / 256), dim3(256), 0, stream>>>(
      Wr, Wn, Wrb);

  // Pass 2: GEMM + router epilogue (same stream -> ordered after pass 1).
  noisy_topk_router_kernel<<<dim3(kRows / kMTile), dim3(256), 0, stream>>>(
      x, Wrb, Wnb, br, bn, eps, out_router, out_idx);
}
